// BidirLSTMLayer_74071005987574
// MI455X (gfx1250) — matrix operand order repaired and checked by reading, not compiled, not run on hardware
//
#include <hip/hip_runtime.h>

// -----------------------------------------------------------------------------
// Bidirectional LSTM for MI455X (gfx1250, wave32, WMMA bf16).
//
//  * 275 GFLOP total but 512 serial steps -> latency bound. Fuse input GEMM
//    into the recurrent GEMM (K = I + H = 1024): no 512 MB x_proj pass.
//  * Packed bf16 weights (8 MB) stay resident in the 192 MB L2 across all
//    512 steps; per-step weight reads are L2 hits.
//  * x pre-converted to bf16 once (parallel kernel) so the serial step loop
//    does zero format conversion -- staging is pure 16B async copies.
//  * Staging uses CDNA5 GLOBAL_LOAD_ASYNC_TO_LDS_B128 (ASYNCcnt) to move
//    global -> LDS without a VGPR round trip.
//  * Per step: grid (32 n-tiles x 2 dirs), 128 threads (4 waves); each wave
//    owns a 16x16 tile for ALL 4 gates (4x v_wmma_f32_16x16x32_bf16 per K
//    chunk), then the fused sigmoid/tanh cell update in-register.
//  * h state ping-pongs between two bf16 buffers (cross-WG RAW safety);
//    c state is element-private and updates in place.
// -----------------------------------------------------------------------------

typedef __attribute__((ext_vector_type(8)))  float  v8f;
typedef __attribute__((ext_vector_type(16))) __bf16 v16bf;

union FragBF {
  uint4 q[2];
  v16bf bf;
};

#define TT   512
#define BB   64
#define II   512
#define HH   512
#define KTOT 1024   // I + H fused K dimension
#define LDA  48     // LDS row stride in shorts (96 B, 16B-aligned, padded)

__device__ __forceinline__ unsigned short f2bf(float f) {
  unsigned int u = __float_as_uint(f);
  u += 0x7FFFu + ((u >> 16) & 1u);      // round-to-nearest-even
  return (unsigned short)(u >> 16);
}
__device__ __forceinline__ float fsigmoid(float x) { return 1.0f / (1.0f + __expf(-x)); }
__device__ __forceinline__ float ftanh(float x)    { return 1.0f - 2.0f / (__expf(2.0f * x) + 1.0f); }

// CDNA5 async global->LDS copy of 16 bytes (per lane). LDS operand is the low
// 32 bits of the generic pointer (flat LDS mapping uses addr[31:0]).
__device__ __forceinline__ void async_copy_b128(void* lds, const void* gaddr) {
  unsigned lds_off = (unsigned)(size_t)lds;
  asm volatile("global_load_async_to_lds_b128 %0, %1, off"
               :: "v"(lds_off), "v"(gaddr) : "memory");
}
__device__ __forceinline__ void wait_async() {
  asm volatile("s_wait_asynccnt 0x0" ::: "memory");
}

// ---------------------------------------------------------------------------
// Prep 1: pack [dir][4H][I | H] weights into bf16 (row-major, K = 1024).
// ---------------------------------------------------------------------------
__global__ void pack_w_kernel(const float* __restrict__ wihf, const float* __restrict__ whhf,
                              const float* __restrict__ wihb, const float* __restrict__ whhb,
                              unsigned short* __restrict__ Wc) {
  int idx = blockIdx.x * 256 + threadIdx.x;          // 2 * 2048 * 1024
  if (idx >= 2 * 2048 * KTOT) return;
  int dir = idx >> 21;
  int rem = idx & ((1 << 21) - 1);
  int row = rem >> 10;
  int col = rem & (KTOT - 1);
  const float* wih = dir ? wihb : wihf;
  const float* whh = dir ? whhb : whhf;
  float v = (col < II) ? wih[row * II + col] : whh[row * HH + (col - II)];
  Wc[idx] = f2bf(v);
}

// ---------------------------------------------------------------------------
// Prep 2: x (f32, T*B*I) -> bf16 once, off the serial critical path.
// ---------------------------------------------------------------------------
__global__ void pack_x_kernel(const float* __restrict__ x, unsigned short* __restrict__ xbf) {
  size_t i = ((size_t)blockIdx.x * 256 + threadIdx.x) * 4;
  if (i >= (size_t)TT * BB * II) return;
  float4 f = *(const float4*)(x + i);
  ushort4 o;
  o.x = f2bf(f.x); o.y = f2bf(f.y); o.z = f2bf(f.z); o.w = f2bf(f.w);
  *(ushort4*)(xbf + i) = o;
}

// ---------------------------------------------------------------------------
// Prep 3: combined bias (b_ih + b_hh), h0 -> bf16 (buffer 0), c0 copy.
// ---------------------------------------------------------------------------
__global__ void init_state_kernel(const float* __restrict__ bihf, const float* __restrict__ bhhf,
                                  const float* __restrict__ bihb, const float* __restrict__ bhhb,
                                  const float* __restrict__ h0f,  const float* __restrict__ h0b,
                                  const float* __restrict__ c0f,  const float* __restrict__ c0b,
                                  float* __restrict__ biasc, unsigned short* __restrict__ hbuf,
                                  float* __restrict__ cst) {
  int idx = blockIdx.x * 256 + threadIdx.x;          // 2 * 64 * 512 = 65536
  if (idx < 2 * 2048) {
    int dir = idx >> 11, r = idx & 2047;
    biasc[idx] = dir ? (bihb[r] + bhhb[r]) : (bihf[r] + bhhf[r]);
  }
  if (idx < 2 * BB * HH) {
    int dir = idx >> 15;
    int rem = idx & (BB * HH - 1);
    const float* h0 = dir ? h0b : h0f;
    const float* c0 = dir ? c0b : c0f;
    hbuf[idx] = f2bf(h0[rem]);                       // ping-pong buffer 0
    cst[idx]  = c0[rem];
  }
}

// ---------------------------------------------------------------------------
// One recurrent step (both directions). grid = (32 n-tiles, 2 dirs), block 128.
// ---------------------------------------------------------------------------
__global__ __launch_bounds__(128) void lstm_step_kernel(
    const unsigned short* __restrict__ xbf, const unsigned short* __restrict__ Wc,
    const float* __restrict__ bias, unsigned short* __restrict__ hbuf,
    float* __restrict__ cst, float* __restrict__ out, int s) {
  const int tid   = threadIdx.x;
  const int lane  = tid & 31;
  const int bt    = tid >> 5;                 // wave id = batch tile (0..3)
  const int n0    = blockIdx.x * 16;          // hidden-column tile base
  const int dir   = blockIdx.y;               // 0 = fwd, 1 = bwd
  const int tpos  = dir ? (TT - 1 - s) : s;   // time index this step touches

  const unsigned short* hin  = hbuf + (size_t)(s & 1) * (2 * BB * HH);
  unsigned short*       hout = hbuf + (size_t)((s + 1) & 1) * (2 * BB * HH);

  __shared__ unsigned short As[BB * LDA];     // 64 x 32 A chunk (batch x K)
  __shared__ unsigned short Bs[BB * LDA];     // 64 x 32 B chunk (4 gates x 16 N x K)

  v8f acc[4] = {v8f{}, v8f{}, v8f{}, v8f{}};

  const int row   = tid >> 1;                 // staging row 0..63
  const int halfc = tid & 1;                  // which 16-col half of the chunk

  // Per-thread B source row: gate = row>>4, n within tile = row&15.
  const size_t wrow = (size_t)dir * (4 * HH) + (size_t)(row >> 4) * HH + n0 + (row & 15);
  const unsigned short* wbase = Wc + wrow * KTOT + halfc * 16;
  const unsigned short* xrow  = xbf + ((size_t)tpos * BB + row) * II + halfc * 16;
  const unsigned short* hrow  = hin + ((size_t)(dir * BB + row)) * HH + halfc * 16;

  for (int kc = 0; kc < KTOT; kc += 32) {
    __syncthreads();   // previous iteration's fragment gathers complete
    // ---- stage A chunk (k < 512 from x_bf16[t], else from h state) + B chunk
    {
      const unsigned short* srcA = (kc < II) ? (xrow + kc) : (hrow + (kc - II));
      unsigned short* dA = &As[row * LDA + halfc * 16];
      async_copy_b128(dA,     srcA);
      async_copy_b128(dA + 8, srcA + 8);
      const unsigned short* srcB = wbase + kc;
      unsigned short* dB = &Bs[row * LDA + halfc * 16];
      async_copy_b128(dB,     srcB);
      async_copy_b128(dB + 8, srcB + 8);
    }
    wait_async();      // this wave's async copies landed in LDS
    __syncthreads();   // all waves' chunks visible

    // ---- gather fragments per ISA 16-bit WMMA layouts, 4 WMMAs (one per gate)
    const int M  = lane & 15;                  // A row / B,C column within tile
    const int hl = lane >> 4;
    FragBF fa;
    const unsigned short* Ar = &As[(bt * 16 + M) * LDA];
    fa.q[0] = *(const uint4*)(Ar + 8 * hl);        // e=0..7 : K = e + 8*hl
    fa.q[1] = *(const uint4*)(Ar + 16 + 8 * hl);   // e=8..15: K = e + 8 + 8*hl
#pragma unroll
    for (int g = 0; g < 4; ++g) {
      FragBF fb;
      const unsigned short* Br = &Bs[(g * 16 + M) * LDA];
      fb.q[0] = *(const uint4*)(Br + 8 * hl);      // e=0..7 : K = e + 8*hl; e=8..15: K = e + 8 + 8*hl
      fb.q[1] = *(const uint4*)(Br + 16 + 8 * hl);
      acc[g] = __builtin_amdgcn_wmma_f32_16x16x32_bf16(
          false, fa.bf, false, fb.bf, (short)0, acc[g], false, false);
    }
  }

  // ---- fused gate nonlinearities + cell update (C/D layout: N=lane&15, M=v+8*hl)
  const int N  = lane & 15;
  const int hl = lane >> 4;
  const int n  = n0 + N;
  const float bI = bias[dir * (4 * HH) + n];
  const float bF = bias[dir * (4 * HH) + HH + n];
  const float bG = bias[dir * (4 * HH) + 2 * HH + n];
  const float bO = bias[dir * (4 * HH) + 3 * HH + n];
#pragma unroll
  for (int v = 0; v < 8; ++v) {
    const int b = bt * 16 + v + 8 * hl;
    const float ig = fsigmoid(acc[0][v] + bI);
    const float fg = fsigmoid(acc[1][v] + bF);
    const float gg = ftanh(acc[2][v] + bG);
    const float og = fsigmoid(acc[3][v] + bO);
    const size_t cidx = (size_t)(dir * BB + b) * HH + n;
    const float cold = cst[cidx];
    const float cnew = fg * cold + ig * gg;
    const float hnew = og * ftanh(cnew);
    cst[cidx]  = cnew;
    hout[cidx] = f2bf(hnew);
    out[((size_t)tpos * BB + b) * (2 * HH) + (size_t)dir * HH + n] = hnew;
    if (s == TT - 1) {  // final carries: outputs | hf | cf | hb | cb
      const size_t base = (size_t)TT * BB * (2 * HH);
      const size_t off  = (size_t)b * HH + n;
      out[base + (size_t)dir * 2 * (BB * HH) + off]           = hnew;
      out[base + (size_t)dir * 2 * (BB * HH) + BB * HH + off] = cnew;
    }
  }
}

// ---------------------------------------------------------------------------
extern "C" void kernel_launch(void* const* d_in, const int* in_sizes, int n_in,
                              void* d_out, int out_size, void* d_ws, size_t ws_size,
                              hipStream_t stream) {
  const float* x     = (const float*)d_in[0];
  const float* h0_f  = (const float*)d_in[1];
  const float* c0_f  = (const float*)d_in[2];
  const float* h0_b  = (const float*)d_in[3];
  const float* c0_b  = (const float*)d_in[4];
  const float* wih_f = (const float*)d_in[5];
  const float* whh_f = (const float*)d_in[6];
  const float* bih_f = (const float*)d_in[7];
  const float* bhh_f = (const float*)d_in[8];
  const float* wih_b = (const float*)d_in[9];
  const float* whh_b = (const float*)d_in[10];
  const float* bih_b = (const float*)d_in[11];
  const float* bhh_b = (const float*)d_in[12];
  float* out = (float*)d_out;

  // workspace layout (~40.6 MB, all 16B-aligned)
  char* ws = (char*)d_ws;
  const size_t W_BYTES    = (size_t)2 * 2048 * KTOT * sizeof(unsigned short); // 8 MB
  const size_t BIAS_BYTES = (size_t)2 * 2048 * sizeof(float);                 // 16 KB
  const size_t H_BYTES    = (size_t)2 * 2 * BB * HH * sizeof(unsigned short); // 256 KB
  const size_t C_BYTES    = (size_t)2 * BB * HH * sizeof(float);              // 256 KB
  unsigned short* Wc    = (unsigned short*)(ws);
  float*          biasc = (float*)(ws + W_BYTES);
  unsigned short* hbuf  = (unsigned short*)(ws + W_BYTES + BIAS_BYTES);
  float*          cst   = (float*)(ws + W_BYTES + BIAS_BYTES + H_BYTES);
  unsigned short* xbf   = (unsigned short*)(ws + W_BYTES + BIAS_BYTES + H_BYTES + C_BYTES);

  pack_w_kernel<<<(2 * 2048 * KTOT + 255) / 256, 256, 0, stream>>>(
      wih_f, whh_f, wih_b, whh_b, Wc);
  pack_x_kernel<<<((TT * BB * II / 4) + 255) / 256, 256, 0, stream>>>(x, xbf);
  init_state_kernel<<<(2 * BB * HH + 255) / 256, 256, 0, stream>>>(
      bih_f, bhh_f, bih_b, bhh_b, h0_f, h0_b, c0_f, c0_b, biasc, hbuf, cst);

  dim3 grid(HH / 16, 2, 1);   // 32 hidden tiles x 2 directions
  for (int s = 0; s < TT; ++s) {
    lstm_step_kernel<<<grid, 128, 0, stream>>>(xbf, Wc, biasc, hbuf, cst, out, s);
  }
}
